// CDSSMBlock_68539088109644
// MI455X (gfx1250) — matrix operand order repaired and checked by reading, not compiled, not run on hardware
//
#include <hip/hip_runtime.h>
#include <hip/hip_bf16.h>
#include <math.h>

// ---------------- problem constants ----------------
#define BB 2
#define LL 1024
#define DMODEL 512
#define DINNER 1024
#define HH 8
#define HD 128
#define NN 32
#define NHALF 16
#define GG 37            // N + 5
#define NGATE (HH*GG)    // 296
#define NPROJ (3*DINNER + HH*NN)  // 3328
#define ROWS (BB*LL)     // 2048

typedef __attribute__((ext_vector_type(16))) _Float16 v16h;
typedef __attribute__((ext_vector_type(8)))  _Float16 v8h;
typedef __attribute__((ext_vector_type(4)))  _Float16 v4h;
typedef __attribute__((ext_vector_type(8)))  float    v8f;

__device__ __forceinline__ float softplusf(float x) {
  return (x > 20.f) ? x : log1pf(expf(x));
}
__device__ __forceinline__ float sigmoidf_(float x) { return 1.f / (1.f + expf(-x)); }
__device__ __forceinline__ float wred32(float v) {
#pragma unroll
  for (int m = 16; m >= 1; m >>= 1) v += __shfl_xor(v, m, 32);
  return v;
}

// ---------------------------------------------------------------------------
// f32 -> f16 conversion pass (vectorized 4-wide), n must be a multiple of 4
// ---------------------------------------------------------------------------
__global__ void cvt_f16_kernel(const float* __restrict__ in,
                               _Float16* __restrict__ out, int n4) {
  int i = blockIdx.x * blockDim.x + threadIdx.x;
  if (i < n4) {
    const float4 v = ((const float4*)in)[i];
    v4h o;
    o[0] = (_Float16)v.x; o[1] = (_Float16)v.y;
    o[2] = (_Float16)v.z; o[3] = (_Float16)v.w;
    ((v4h*)out)[i] = o;
  }
}

// ---------------------------------------------------------------------------
// C[M,N] = A[M,K] @ W[N,K]^T (+bias) (+residual R[M,N]) via
// v_wmma_f32_16x16x32_f16. A/W are pre-converted f16; fragments load as
// global_load_b128 (8 halves). One wave computes a 16x64 strip (4 accums
// reusing one A fragment). Requires M%16==0, K%32==0, 16B-aligned buffers.
// A lane layout (16-bit 16x32): half h holds K {8h..8h+7} U {16+8h..23+8h}.
// B lane layout (32x16): half h holds K {8h..8h+7} U {16+8h..23+8h}.
// ---------------------------------------------------------------------------
__global__ void __launch_bounds__(128)
gemm_f16wmma(const _Float16* __restrict__ A, const _Float16* __restrict__ W,
             const float* __restrict__ bias, const float* __restrict__ R,
             float* __restrict__ C, int M, int N, int K) {
  const int wave = threadIdx.x >> 5;
  const int lane = threadIdx.x & 31;
  const int halfId = lane >> 4;
  const int idx  = lane & 15;

  const int tm  = M >> 4;
  const int tn4 = (N + 63) >> 6;
  long tile = (long)blockIdx.x * 4 + wave;
  if (tile >= (long)tm * tn4) return;
  const int tile_m = (int)(tile / tn4);
  const int ncol0 = (int)(tile % tn4) * 64;

  const _Float16* Ap = A + (long)(tile_m * 16 + idx) * K + 8 * halfId;
  const int kb0 = 8 * halfId;

  v8f acc[4];
#pragma unroll
  for (int s = 0; s < 4; ++s)
#pragma unroll
    for (int j = 0; j < 8; ++j) acc[s][j] = 0.f;

  if (ncol0 + 64 <= N) {
    // ---- fast path: full 16x64 strip, no masking ----
    const _Float16* Wp[4];
#pragma unroll
    for (int s = 0; s < 4; ++s)
      Wp[s] = W + (long)(ncol0 + s * 16 + idx) * K + kb0;
    for (int k = 0; k < K; k += 32) {
      v16h af;
      *(v8h*)&af       = *(const v8h*)(Ap + k);
      *((v8h*)&af + 1) = *(const v8h*)(Ap + k + 16);
#pragma unroll
      for (int s = 0; s < 4; ++s) {
        v16h bf;
        *(v8h*)&bf       = *(const v8h*)(Wp[s] + k);
        *((v8h*)&bf + 1) = *(const v8h*)(Wp[s] + k + 16);
        acc[s] = __builtin_amdgcn_wmma_f32_16x16x32_f16(
            false, af, false, bf, (short)0, acc[s], false, false);
      }
    }
  } else {
    // ---- edge path: clamp invalid W rows to row 0 and zero the fragment ----
    const _Float16* Wp[4];
    _Float16 wm[4];
#pragma unroll
    for (int s = 0; s < 4; ++s) {
      int rw = ncol0 + s * 16 + idx;
      wm[s] = (rw < N) ? (_Float16)1.f : (_Float16)0.f;
      Wp[s] = W + (long)((rw < N) ? rw : 0) * K + kb0;
    }
    for (int k = 0; k < K; k += 32) {
      v16h af;
      *(v8h*)&af       = *(const v8h*)(Ap + k);
      *((v8h*)&af + 1) = *(const v8h*)(Ap + k + 16);
#pragma unroll
      for (int s = 0; s < 4; ++s) {
        v16h bf;
        *(v8h*)&bf       = *(const v8h*)(Wp[s] + k);
        *((v8h*)&bf + 1) = *(const v8h*)(Wp[s] + k + 16);
        bf = bf * wm[s];
        acc[s] = __builtin_amdgcn_wmma_f32_16x16x32_f16(
            false, af, false, bf, (short)0, acc[s], false, false);
      }
    }
  }

#pragma unroll
  for (int s = 0; s < 4; ++s) {
    int col = ncol0 + s * 16 + idx;
    if (col < N) {
      float bv = bias ? bias[col] : 0.f;
#pragma unroll
      for (int j = 0; j < 8; ++j) {
        long rowc = (long)tile_m * 16 + j + 8 * halfId;
        float v = acc[s][j] + bv;
        if (R) v += R[rowc * N + col];
        C[rowc * N + col] = v;
      }
    }
  }
}

// ---------------------------------------------------------------------------
// RMSNorm: one wave per row of 512, 8 rows per 256-thread block.
// Output only consumed by the in_proj GEMM -> emit f16 directly.
// ---------------------------------------------------------------------------
__global__ void rmsnorm_kernel(const float* __restrict__ x,
                               const float* __restrict__ w,
                               _Float16* __restrict__ out16) {
  const int wave = threadIdx.x >> 5, lane = threadIdx.x & 31;
  const long row = (long)blockIdx.x * 8 + wave;
  const float* xr = x + row * DMODEL;
  float v[16], ss = 0.f;
#pragma unroll
  for (int j = 0; j < 16; ++j) { v[j] = xr[lane + 32 * j]; ss += v[j] * v[j]; }
  ss = wred32(ss);
  const float r = rsqrtf(ss / (float)DMODEL + 1e-5f);
  _Float16* orow = out16 + row * DMODEL;
#pragma unroll
  for (int j = 0; j < 16; ++j)
    orow[lane + 32 * j] = (_Float16)(v[j] * r * w[lane + 32 * j]);
}

// ---------------------------------------------------------------------------
// Causal depthwise conv (K=4) + bias + SiLU. Input: proj[:, DINNER:2*DINNER].
// Writes f32 (needed by Dskip fuse) and f16 (GEMM operand).
// ---------------------------------------------------------------------------
__global__ void conv_silu_kernel(const float* __restrict__ proj,
                                 const float* __restrict__ cw,
                                 const float* __restrict__ cb,
                                 float* __restrict__ xc,
                                 _Float16* __restrict__ xc16) {
  long gid = (long)blockIdx.x * blockDim.x + threadIdx.x;
  if (gid >= (long)ROWS * DINNER) return;
  const int c = (int)(gid & (DINNER - 1));
  const int l = (int)((gid >> 10) & (LL - 1));
  const int b = (int)(gid >> 20);
  float acc = cb[c];
#pragma unroll
  for (int j = 0; j < 4; ++j) {
    int ll = l + j - 3;
    if (ll >= 0)
      acc += proj[((long)b * LL + ll) * NPROJ + DINNER + c] * cw[c * 4 + j];
  }
  const float s = acc / (1.f + expf(-acc));   // silu
  xc[gid] = s;
  xc16[gid] = (_Float16)s;
}

// ---------------------------------------------------------------------------
// Per-(b,l): L2-norm K and Q (Q in place), qk dot, gate dynamics, V mixing.
// 256 threads = 8 waves, wave h owns head h.
// ---------------------------------------------------------------------------
__global__ void prep_kernel(const float* __restrict__ proj, float* __restrict__ qbuf,
                            const float* __restrict__ v_first, const float* __restrict__ gates,
                            const float* __restrict__ log_dt_scale,
                            const float* __restrict__ v_res_gate,
                            float* __restrict__ kn, float* __restrict__ vg,
                            float* __restrict__ are, float* __restrict__ aim,
                            float* __restrict__ betab, float* __restrict__ selcb,
                            float* __restrict__ qkb) {
  const long row = blockIdx.x;
  const int h = threadIdx.x >> 5, lane = threadIdx.x & 31;

  const float* kp = proj + row * NPROJ + 2 * DINNER + h * HD;
  float kv[4], sk = 0.f;
#pragma unroll
  for (int j = 0; j < 4; ++j) { kv[j] = kp[lane + 32 * j]; sk += kv[j] * kv[j]; }
  sk = wred32(sk);
  const float rk = rsqrtf(sk + 1e-6f);

  float* qp = qbuf + row * DINNER + h * HD;
  float qv[4], sq = 0.f;
#pragma unroll
  for (int j = 0; j < 4; ++j) { qv[j] = qp[lane + 32 * j]; sq += qv[j] * qv[j]; }
  sq = wred32(sq);
  const float rq = rsqrtf(sq + 1e-6f);

  float* ko = kn + row * DINNER + h * HD;
  float qk = 0.f;
#pragma unroll
  for (int j = 0; j < 4; ++j) {
    float knj = kv[j] * rk, qnj = qv[j] * rq;
    ko[lane + 32 * j] = knj;
    qp[lane + 32 * j] = qnj;
    qk += knj * qnj;
  }
  qk = wred32(qk);
  if (lane == 0) qkb[row * HH + h] = qk;

  const float* g = gates + row * NGATE + h * GG;
  if (lane < NHALF) {
    const int n = lane;
    const float dt = softplusf(g[34] + log_dt_scale[h]) + 1e-3f;
    const float freq = expf(-((float)h * 0.125f) * logf(10000.f));
    const float lam_re = -softplusf(g[n]);
    const float lam_im = g[NHALF + n] + freq;
    const float hdt = 0.5f * dt;
    const float nr = 1.f + hdt * lam_re, ni = hdt * lam_im;
    const float dr = 1.f - hdt * lam_re, di = -hdt * lam_im;
    const float den = dr * dr + di * di;
    const float r = sigmoidf_(g[36]);
    const float ar = (nr * dr + ni * di) / den * r;
    const float ai = (ni * dr - nr * di) / den * r;
    const float vp = sqrtf(fmaxf(1.f - (ar * ar + ai * ai), 1e-6f));
    are[row * (HH * NHALF) + h * NHALF + n] = ar;
    aim[row * (HH * NHALF) + h * NHALF + n] = ai;
    const float nu = sigmoidf_(v_res_gate[h]);
#pragma unroll
    for (int c = 0; c < 2; ++c) {
      int jj = 2 * n + c;
      float v  = proj[row * NPROJ + 3 * DINNER + h * NN + jj];
      float vf = v_first[row * (HH * NN) + h * NN + jj];
      vg[row * (HH * NN) + h * NN + jj] = (vf + nu * (v - vf)) * vp;
    }
    if (n == 0) {
      betab[row * HH + h] = sigmoidf_(g[35]) * sigmoidf_(g[32]);
      selcb[row * HH + h] = sigmoidf_(g[33]);
    }
  }
}

// ---------------------------------------------------------------------------
// Sequential delta-rule scan fused with retrieval (Y never materialized).
// One block per (b,h): 256 threads = 16 n-rows x 16 d-slices; state in regs.
// Output only feeds the readout GEMM -> emit f16.
// ---------------------------------------------------------------------------
__global__ void __launch_bounds__(256)
scan_kernel(const float* __restrict__ kn, const float* __restrict__ qn,
            const float* __restrict__ vg, const float* __restrict__ are,
            const float* __restrict__ aim, const float* __restrict__ betab,
            const float* __restrict__ selcb, const float* __restrict__ qkb,
            const float* __restrict__ ssc, _Float16* __restrict__ retr16) {
  const int b = blockIdx.x >> 3, h = blockIdx.x & 7;
  const int t = threadIdx.x;
  const int n = t >> 4, i = t & 15;
  __shared__ float k_s[HD], q_s[HD], ar_s[NHALF], ai_s[NHALF], vr_s[NHALF], vi_s[NHALF];
  __shared__ float sc_s[3];  // beta, selC, qk
  const float ss = ssc[0];

  float sre[8], sim[8];
#pragma unroll
  for (int j = 0; j < 8; ++j) { sre[j] = 0.f; sim[j] = 0.f; }

  for (int step = 0; step < LL; ++step) {
    const long row = (long)b * LL + step;
    if (t < 128) {
      k_s[t] = kn[row * DINNER + h * HD + t];
      q_s[t] = qn[row * DINNER + h * HD + t];
    } else if (t < 144) ar_s[t - 128] = are[row * (HH * NHALF) + h * NHALF + (t - 128)];
    else if (t < 160) ai_s[t - 144] = aim[row * (HH * NHALF) + h * NHALF + (t - 144)];
    else if (t < 176) vr_s[t - 160] = vg[row * (HH * NN) + h * NN + 2 * (t - 160)];
    else if (t < 192) vi_s[t - 176] = vg[row * (HH * NN) + h * NN + 2 * (t - 176) + 1];
    else if (t == 192) sc_s[0] = betab[row * HH + h];
    else if (t == 193) sc_s[1] = selcb[row * HH + h];
    else if (t == 194) sc_s[2] = qkb[row * HH + h];
    __syncthreads();

    const float ar = ar_s[n], ai = ai_s[n];
    float pr = 0.f, pi = 0.f;
    float kd[8];
#pragma unroll
    for (int j = 0; j < 8; ++j) {
      const int d = i + 16 * j;
      kd[j] = k_s[d];
      const float r0 = ar * sre[j] - ai * sim[j];
      const float i0 = ar * sim[j] + ai * sre[j];
      sre[j] = r0; sim[j] = i0;
      pr += r0 * kd[j]; pi += i0 * kd[j];
    }
#pragma unroll
    for (int m = 8; m >= 1; m >>= 1) {      // reduce within 16-lane half
      pr += __shfl_xor(pr, m, 32);
      pi += __shfl_xor(pi, m, 32);
    }
    const float beta = sc_s[0];
    const float er = (vr_s[n] - pr) * beta;
    const float ei = (vi_s[n] - pi) * beta;
    float rr = 0.f, ri = 0.f;
#pragma unroll
    for (int j = 0; j < 8; ++j) {
      const int d = i + 16 * j;
      sre[j] += er * kd[j];
      sim[j] += ei * kd[j];
      rr += sre[j] * q_s[d];
      ri += sim[j] * q_s[d];
    }
#pragma unroll
    for (int m = 8; m >= 1; m >>= 1) {
      rr += __shfl_xor(rr, m, 32);
      ri += __shfl_xor(ri, m, 32);
    }
    if (i == 0) {
      const float qk = sc_s[2], sel = sc_s[1];
      const float sh = ss * qk * qk;
      retr16[row * (HH * NN) + h * NN + 2 * n]     = (_Float16)((rr + sh * vr_s[n]) * sel);
      retr16[row * (HH * NN) + h * NN + 2 * n + 1] = (_Float16)((ri + sh * vi_s[n]) * sel);
    }
    __syncthreads();
  }
}

// ---------------------------------------------------------------------------
// GroupNorm stats: per (b, group) over 128 channels x all L (131072 elems)
// ---------------------------------------------------------------------------
__global__ void gn_reduce_kernel(const float* __restrict__ y, float* __restrict__ stats) {
  const int b = blockIdx.x >> 3, g = blockIdx.x & 7;
  float s = 0.f, sq = 0.f;
  for (int idx = threadIdx.x; idx < LL * 128; idx += blockDim.x) {
    const int l = idx >> 7, c = idx & 127;
    const float v = y[((long)b * LL + l) * DINNER + g * 128 + c];
    s += v; sq += v * v;
  }
  s = wred32(s); sq = wred32(sq);
  __shared__ float bs[8], bq[8];
  const int wave = threadIdx.x >> 5, lane = threadIdx.x & 31;
  if (lane == 0) { bs[wave] = s; bq[wave] = sq; }
  __syncthreads();
  if (threadIdx.x == 0) {
    float S = 0.f, Q = 0.f;
#pragma unroll
    for (int w = 0; w < 8; ++w) { S += bs[w]; Q += bq[w]; }
    const float cnt = (float)(LL * 128);
    const float mean = S / cnt;
    const float var = Q / cnt - mean * mean;
    stats[blockIdx.x * 2]     = mean;
    stats[blockIdx.x * 2 + 1] = rsqrtf(var + 1e-5f);
  }
}

// ---------------------------------------------------------------------------
// GroupNorm affine + *silu(z) + Dskip*x_conv. Output feeds out GEMM -> f16.
// ---------------------------------------------------------------------------
__global__ void fuse_kernel(const float* __restrict__ y, const float* __restrict__ stats,
                            const float* __restrict__ gw, const float* __restrict__ gb,
                            const float* __restrict__ proj, const float* __restrict__ xc,
                            const float* __restrict__ Dskip, _Float16* __restrict__ yfin16) {
  long gid = (long)blockIdx.x * blockDim.x + threadIdx.x;
  if (gid >= (long)ROWS * DINNER) return;
  const int c = (int)(gid & (DINNER - 1));
  const long row = gid >> 10;
  const int b = (int)(row >> 10);
  const int g = c >> 7;
  const float mean = stats[(b * 8 + g) * 2];
  const float rstd = stats[(b * 8 + g) * 2 + 1];
  const float yn = (y[gid] - mean) * rstd * gw[c] + gb[c];
  const float z = proj[row * NPROJ + c];
  const float sz = z / (1.f + expf(-z));
  yfin16[gid] = (_Float16)(yn * sz + Dskip[c] * xc[gid]);
}

// ---------------------------------------------------------------------------
static void launch_gemm(const _Float16* A, const _Float16* W, const float* bias,
                        const float* R, float* C, int M, int N, int K,
                        hipStream_t s) {
  const int tm = M >> 4;
  const int tn4 = (N + 63) >> 6;
  const long tiles = (long)tm * tn4;
  const int blocks = (int)((tiles + 3) / 4);
  gemm_f16wmma<<<blocks, 128, 0, s>>>(A, W, bias, R, C, M, N, K);
}

static void launch_cvt(const float* in, _Float16* out, long n, hipStream_t s) {
  int n4 = (int)(n / 4);
  cvt_f16_kernel<<<(n4 + 255) / 256, 256, 0, s>>>(in, out, n4);
}

extern "C" void kernel_launch(void* const* d_in, const int* in_sizes, int n_in,
                              void* d_out, int out_size, void* d_ws, size_t ws_size,
                              hipStream_t stream) {
  const float* x       = (const float*)d_in[0];
  const float* v_first = (const float*)d_in[1];
  const float* norm_w  = (const float*)d_in[2];
  const float* in_w    = (const float*)d_in[3];
  const float* in_b    = (const float*)d_in[4];
  const float* conv_w  = (const float*)d_in[5];
  const float* conv_b  = (const float*)d_in[6];
  const float* gate_w  = (const float*)d_in[7];
  const float* gate_b  = (const float*)d_in[8];
  const float* log_dt  = (const float*)d_in[9];
  const float* q_w     = (const float*)d_in[10];
  const float* read_w  = (const float*)d_in[11];
  const float* out_w   = (const float*)d_in[12];
  const float* gn_w    = (const float*)d_in[13];
  const float* gn_b    = (const float*)d_in[14];
  const float* Dskip   = (const float*)d_in[15];
  const float* vrg     = (const float*)d_in[16];
  const float* ssc     = (const float*)d_in[17];
  float* out = (float*)d_out;

  // -------- f32 scratch --------
  float* ws = (float*)d_ws;
  float* proj  = ws;                                 // 2048*3328
  float* xconv = proj  + (long)ROWS * NPROJ;         // 2048*1024
  float* gates = xconv + (long)ROWS * DINNER;        // 2048*296
  float* qbuf  = gates + (long)ROWS * NGATE;         // 2048*1024
  float* kn    = qbuf  + (long)ROWS * DINNER;        // 2048*1024
  float* vg    = kn    + (long)ROWS * DINNER;        // 2048*256
  float* are   = vg    + (long)ROWS * HH * NN;       // 2048*128
  float* aim   = are   + (long)ROWS * HH * NHALF;    // 2048*128
  float* betab = aim   + (long)ROWS * HH * NHALF;    // 2048*8
  float* selcb = betab + (long)ROWS * HH;            // 2048*8
  float* qkb   = selcb + (long)ROWS * HH;            // 2048*8
  float* ybuf  = qkb   + (long)ROWS * HH;            // 2048*1024
  float* stats = ybuf  + (long)ROWS * DINNER;        // 32
  // -------- f16 scratch (16B aligned: all sizes are multiples of 8) --------
  _Float16* h16   = (_Float16*)(stats + 32);
  _Float16* xn16    = h16;                                   // 2048*512
  _Float16* xconv16 = xn16    + (long)ROWS * DMODEL;         // 2048*1024
  _Float16* retr16  = xconv16 + (long)ROWS * DINNER;         // 2048*256
  _Float16* yfin16  = retr16  + (long)ROWS * HH * NN;        // 2048*1024
  _Float16* w_in16  = yfin16  + (long)ROWS * DINNER;         // 3328*512
  _Float16* w_gt16  = w_in16  + (long)NPROJ * DMODEL;        // 296*1024
  _Float16* w_q16   = w_gt16  + (long)NGATE * DINNER;        // 1024*1024
  _Float16* w_rd16  = w_q16   + (long)DINNER * DINNER;       // 1024*256
  _Float16* w_out16 = w_rd16  + (long)DINNER * (HH * NN);    // 512*1024

  // 0. one-shot weight conversions to f16
  launch_cvt(in_w,   w_in16,  (long)NPROJ * DMODEL,      stream);
  launch_cvt(gate_w, w_gt16,  (long)NGATE * DINNER,      stream);
  launch_cvt(q_w,    w_q16,   (long)DINNER * DINNER,     stream);
  launch_cvt(read_w, w_rd16,  (long)DINNER * (HH * NN),  stream);
  launch_cvt(out_w,  w_out16, (long)DMODEL * DINNER,     stream);

  // 1. RMSNorm -> f16
  rmsnorm_kernel<<<ROWS / 8, 256, 0, stream>>>(x, norm_w, xn16);
  // 2. in_proj GEMM (2048 x 3328, K=512)
  launch_gemm(xn16, w_in16, in_b, nullptr, proj, ROWS, NPROJ, DMODEL, stream);
  // 3. causal depthwise conv + SiLU (f32 + f16 outputs)
  conv_silu_kernel<<<(ROWS * DINNER) / 256, 256, 0, stream>>>(proj, conv_w, conv_b,
                                                              xconv, xconv16);
  // 4. gate GEMM (2048 x 296, K=1024)
  launch_gemm(xconv16, w_gt16, gate_b, nullptr, gates, ROWS, NGATE, DINNER, stream);
  // 5. Q GEMM (2048 x 1024, K=1024)
  launch_gemm(xconv16, w_q16, nullptr, nullptr, qbuf, ROWS, DINNER, DINNER, stream);
  // 6. dynamics + L2 norms + V mixing + qk
  prep_kernel<<<ROWS, 256, 0, stream>>>(proj, qbuf, v_first, gates, log_dt, vrg,
                                        kn, vg, are, aim, betab, selcb, qkb);
  // 7. fused scan + retrieval (Y never materialized) -> f16
  scan_kernel<<<BB * HH, 256, 0, stream>>>(kn, qbuf, vg, are, aim, betab, selcb,
                                           qkb, ssc, retr16);
  // 8. readout GEMM (2048 x 1024, K=256)
  launch_gemm(retr16, w_rd16, nullptr, nullptr, ybuf, ROWS, DINNER, HH * NN, stream);
  // 9-10. GroupNorm (stats over group x full sequence) + gating fuse -> f16
  gn_reduce_kernel<<<BB * 8, 256, 0, stream>>>(ybuf, stats);
  fuse_kernel<<<(ROWS * DINNER) / 256, 256, 0, stream>>>(ybuf, stats, gn_w, gn_b,
                                                         proj, xconv, Dskip, yfin16);
  // 11. out GEMM (2048 x 512, K=1024) with residual add
  launch_gemm(yfin16, w_out16, nullptr, x, out, ROWS, DMODEL, DINNER, stream);
}
